// BuNNHopLayer_5875515261231
// MI455X (gfx1250) — matrix operand order repaired and checked by reading, not compiled, not run on hardware
//
#include <hip/hip_runtime.h>
#include <math.h>

typedef __attribute__((ext_vector_type(16))) _Float16 v16h;
typedef __attribute__((ext_vector_type(4)))  _Float16 v4h;
typedef __attribute__((ext_vector_type(8)))  float    v8f;

#define C_DIM 128
#define NB_DIM 32

// WMMA 16x32 f16 A-fragment addressing (ISA 7.12.2):
// lane = half*16 + (row%16); elem for K kk (0..31):
//   half = (kk>>3)&1 ; kp = kk-8*half in {0..7,16..23}
//   p = kp<8 ? kp/2 : 4+(kp-16)/2 ; elem = 2p + (kk&1)
__device__ __forceinline__ size_t apack_off(int n, int k) {
  int chunk = k >> 5, kk = k & 31;
  int half  = (kk >> 3) & 1;
  int kp    = kk - 8 * half;
  int p     = (kp < 8) ? (kp >> 1) : (4 + ((kp - 16) >> 1));
  int elem  = 2 * p + (kk & 1);
  return ((((size_t)(n >> 4) * 4 + chunk) * 32) + (size_t)(half * 16 + (n & 15))) * 16 + elem;
}

// ---------------- degree / normalization ----------------
__global__ void k_deg_init(float* deg, int N) {
  int i = blockIdx.x * blockDim.x + threadIdx.x;
  if (i < N) deg[i] = 1.0f;  // self loop counts once
}
__global__ void k_deg_count(const long long* __restrict__ src, float* deg, int E) {
  int e = blockIdx.x * blockDim.x + threadIdx.x;
  if (e < E) atomicAdd(&deg[(int)src[e]], 1.0f);
}
__global__ void k_deg_inv(float* deg, int N) {
  int i = blockIdx.x * blockDim.x + threadIdx.x;
  if (i < N) deg[i] = 1.0f / deg[i];
}

// ---------------- struct_enc MLP -> angles; rotate x; pack f16 A-fragments ----------
// one wave32 per node; lane b owns bundle b (4 channels = 4 consecutive frag elems)
__global__ void k_angles_rotate(const float* __restrict__ x,
                                const float* __restrict__ w1, const float* __restrict__ b1,
                                const float* __restrict__ w2, const float* __restrict__ b2,
                                float* __restrict__ cth, float* __restrict__ sth,
                                _Float16* __restrict__ Apack, int N) {
  __shared__ float smid[8][64];
  int wave = threadIdx.x >> 5;
  int lane = threadIdx.x & 31;
  int n = blockIdx.x * 8 + wave;
  bool act = (n < N);
  const float* xr = x + (size_t)n * C_DIM;
  if (act) {
    for (int jj = 0; jj < 2; ++jj) {
      int j = lane + 32 * jj;
      float a = b1[j];
      for (int k = 0; k < C_DIM; ++k) a = fmaf(xr[k], w1[k * 64 + j], a);
      smid[wave][j] = 0.5f * a * (1.0f + erff(a * 0.70710678f));  // exact gelu
    }
  }
  __syncthreads();
  if (!act) return;
  float a = b2[lane];
  for (int j = 0; j < 64; ++j) a = fmaf(smid[wave][j], w2[j * NB_DIM + lane], a);
  float th = 3.14159265358979f * tanhf(a);
  float cv = cosf(th), sv = sinf(th);
  cth[(size_t)n * NB_DIM + lane] = cv;
  sth[(size_t)n * NB_DIM + lane] = sv;
  // vf[n,b] = [[v00,v01],[v10,v11]];  R @ vf, stored straight into A-fragment layout
  float v00 = xr[lane*4+0], v01 = xr[lane*4+1], v10 = xr[lane*4+2], v11 = xr[lane*4+3];
  v4h r;
  r.x = (_Float16)(cv*v00 - sv*v10);
  r.y = (_Float16)(cv*v01 - sv*v11);
  r.z = (_Float16)(sv*v00 + cv*v10);
  r.w = (_Float16)(sv*v01 + cv*v11);
  *(v4h*)(Apack + apack_off(n, lane * 4)) = r;   // 4 consecutive elems, 8B aligned
}

// pack lin_w (f32 row-major [K=128][N=128]) into WMMA B-fragment layout, f16
// B 32x16 layout: lane = half*16 + col%16 ; half = (kk>>3)&1, elem = (kk&7) + 8*(kk>>4) (kk = K % 32)
__global__ void k_packB(const float* __restrict__ w, _Float16* __restrict__ Bpack) {
  int i = blockIdx.x * blockDim.x + threadIdx.x;   // i = k*128 + n
  if (i >= C_DIM * C_DIM) return;
  int k = i >> 7, n = i & 127;
  int ctile = n >> 4, sub = n & 15;
  int chunk = k >> 5, kk = k & 31;
  int half = (kk >> 3) & 1, elem = (kk & 7) + ((kk >> 4) << 3);
  size_t off = ((((size_t)ctile * 4 + chunk) * 32) + (size_t)(half * 16 + sub)) * 16 + elem;
  Bpack[off] = (_Float16)w[i];
}

__global__ void k_att_softmax(const float* __restrict__ att, float* __restrict__ att_s) {
  int b = threadIdx.x;
  if (b >= NB_DIM) return;
  float v0 = att[b], v1 = att[NB_DIM + b], v2 = att[2*NB_DIM + b], v3 = att[3*NB_DIM + b];
  float m = fmaxf(fmaxf(v0, v1), fmaxf(v2, v3));
  float e0 = expf(v0 - m), e1 = expf(v1 - m), e2 = expf(v2 - m), e3 = expf(v3 - m);
  float s = 1.0f / (e0 + e1 + e2 + e3);
  att_s[b] = e0*s; att_s[NB_DIM+b] = e1*s; att_s[2*NB_DIM+b] = e2*s; att_s[3*NB_DIM+b] = e3*s;
}

// ---------------- h = rotx @ lin_w + lin_b via v_wmma_f32_16x16x32_f16 ----------------
// block = 256 threads = 8 waves; wave w owns column tile w; fragments are pre-packed,
// so each K-chunk is 2x b128 loads per operand + one WMMA.
__global__ void k_gemm_wmma(const _Float16* __restrict__ Apack,
                            const _Float16* __restrict__ Bpack,
                            const float* __restrict__ bias,
                            float* __restrict__ out, int N) {
  int wave = threadIdx.x >> 5;
  int lane = threadIdx.x & 31;
  int m0 = blockIdx.x * 16;
  int n0 = wave * 16;
  int sub = lane & 15, half = lane >> 4;

  const v16h* Af = (const v16h*)Apack + (size_t)blockIdx.x * 4 * 32 + lane;
  const v16h* Bf = (const v16h*)Bpack + (size_t)wave     * 4 * 32 + lane;

  v8f acc = {};
#pragma unroll
  for (int c = 0; c < 4; ++c) {
    v16h a = Af[c * 32];
    v16h b = Bf[c * 32];
    acc = __builtin_amdgcn_wmma_f32_16x16x32_f16(false, a, false, b, (short)0, acc,
                                                 false, false);
  }
  int col = n0 + sub;
  float bv = bias[col];
  if (m0 + 16 <= N) {
    // full tile: wave-uniform branch, unguarded stores (common path)
    float* op = out + (size_t)(m0 + 8 * half) * C_DIM + col;
#pragma unroll
    for (int r = 0; r < 8; ++r) op[(size_t)r * C_DIM] = acc[r] + bv;
  } else {
#pragma unroll
    for (int r = 0; r < 8; ++r) {
      int row = m0 + r + 8 * half;               // D: VGPR r -> row r + 8*half
      if (row < N) out[(size_t)row * C_DIM + col] = acc[r] + bv;
    }
  }
}

// ---------------- propagation ----------------
__global__ void k_zero(float* p, int n) {
  int i = blockIdx.x * blockDim.x + threadIdx.x;
  if (i < n) p[i] = 0.0f;
}
// self-loop term initializes the output: out[n] = h[n] * inv_deg[n]
__global__ void k_prop_self(const float* __restrict__ hin, const float* __restrict__ invd,
                            float* __restrict__ hout, int N) {
  int i = blockIdx.x * blockDim.x + threadIdx.x;
  if (i < N * C_DIM) hout[i] = hin[i] * invd[i >> 7];
}
// one wave per edge, lane owns a 4-channel group: out[dst] += h[src] * inv_deg[dst]
__global__ void k_prop_edges(const long long* __restrict__ src, const long long* __restrict__ dst,
                             const float* __restrict__ invd,
                             const float* __restrict__ hin, float* __restrict__ hout, int E) {
  long long tid = (long long)blockIdx.x * blockDim.x + threadIdx.x;
  long long e = tid >> 5;
  if (e >= E) return;
  int cg = ((int)tid & 31) << 2;
  int s = (int)src[e];
  int d = (int)dst[e];
  float w = invd[d];
  const float4 v = *(const float4*)(hin + (size_t)s * C_DIM + cg);
  float* o = hout + (size_t)d * C_DIM + cg;
  atomicAdd(o + 0, v.x * w);
  atomicAdd(o + 1, v.y * w);
  atomicAdd(o + 2, v.z * w);
  atomicAdd(o + 3, v.w * w);
}
// fused attention combine at a checkpoint: acc += att_row[bundle(c)] * h
__global__ void k_accum(float* __restrict__ acc, const float* __restrict__ h,
                        const float* __restrict__ att_row, int N) {
  int i = blockIdx.x * blockDim.x + threadIdx.x;
  if (i < N * C_DIM) acc[i] += att_row[(i & 127) >> 2] * h[i];
}

// ---------------- inverse rotation + residual (in place on acc = d_out) ----------------
__global__ void k_final(const float* __restrict__ x,
                        const float* __restrict__ cth, const float* __restrict__ sth,
                        float* __restrict__ acc, int N) {
  int tid = blockIdx.x * blockDim.x + threadIdx.x;
  if (tid >= N * NB_DIM) return;
  int n = tid >> 5;
  int b = tid & 31;
  float cv = cth[tid], sv = sth[tid];
  const float4 vx = *(const float4*)(x + (size_t)n * C_DIM + b * 4);
  float4 v = *(float4*)(acc + (size_t)n * C_DIM + b * 4);
  // R^T @ v
  float r0 =  cv * v.x + sv * v.z;
  float r1 =  cv * v.y + sv * v.w;
  float r2 = -sv * v.x + cv * v.z;
  float r3 = -sv * v.y + cv * v.w;
  float4 o;
  o.x = vx.x + r0; o.y = vx.y + r1; o.z = vx.z + r2; o.w = vx.w + r3;
  *(float4*)(acc + (size_t)n * C_DIM + b * 4) = o;
}

extern "C" void kernel_launch(void* const* d_in, const int* in_sizes, int n_in,
                              void* d_out, int out_size, void* d_ws, size_t ws_size,
                              hipStream_t stream) {
  const float*     x         = (const float*)d_in[0];
  const long long* ei        = (const long long*)d_in[1];   // int64 [2,E]
  const float*     lin_w     = (const float*)d_in[2];
  const float*     lin_b     = (const float*)d_in[3];
  const float*     enc_w1    = (const float*)d_in[4];
  const float*     enc_b1    = (const float*)d_in[5];
  const float*     enc_w2    = (const float*)d_in[6];
  const float*     enc_b2    = (const float*)d_in[7];
  const float*     attention = (const float*)d_in[8];

  const int N = in_sizes[0] / C_DIM;
  const int E = in_sizes[1] / 2;
  const long long* src = ei;
  const long long* dst = ei + E;

  // carve workspace (256B aligned slices)
  char* wp = (char*)d_ws;
  auto carve = [&](size_t bytes) -> char* {
    char* p = wp;
    wp += (bytes + 255) & ~(size_t)255;
    return p;
  };
  const size_t ntiles = (size_t)(N + 15) / 16;
  float*    deg   = (float*)carve(sizeof(float) * (size_t)N);
  float*    cth   = (float*)carve(sizeof(float) * (size_t)N * NB_DIM);
  float*    sth   = (float*)carve(sizeof(float) * (size_t)N * NB_DIM);
  _Float16* Apack = (_Float16*)carve(sizeof(_Float16) * ntiles * 4 * 32 * 16);
  _Float16* Bpack = (_Float16*)carve(sizeof(_Float16) * C_DIM * C_DIM);
  float*    att_s = (float*)carve(sizeof(float) * 128);
  float*    hA    = (float*)carve(sizeof(float) * (size_t)N * C_DIM);
  float*    hB    = (float*)carve(sizeof(float) * (size_t)N * C_DIM);
  float*    acc   = (float*)d_out;                 // attention accumulator lives in d_out

  const int TPB = 256;
  const int NC  = N * C_DIM;

  // normalization
  k_deg_init <<<(N + TPB - 1) / TPB, TPB, 0, stream>>>(deg, N);
  k_deg_count<<<(E + TPB - 1) / TPB, TPB, 0, stream>>>(src, deg, E);
  k_deg_inv  <<<(N + TPB - 1) / TPB, TPB, 0, stream>>>(deg, N);

  // angles + rotation (writes A-fragments); weight pack; attention softmax
  k_angles_rotate<<<(N + 7) / 8, 256, 0, stream>>>(x, enc_w1, enc_b1, enc_w2, enc_b2,
                                                   cth, sth, Apack, N);
  k_packB      <<<(C_DIM * C_DIM + TPB - 1) / TPB, TPB, 0, stream>>>(lin_w, Bpack);
  k_att_softmax<<<1, 32, 0, stream>>>(attention, att_s);

  // dense linear via WMMA: hA = rot(x) @ lin_w + lin_b
  k_gemm_wmma<<<(int)ntiles, 256, 0, stream>>>(Apack, Bpack, lin_b, hA, N);

  // multi-hop propagation with fused checkpoint combine
  k_zero<<<(NC + TPB - 1) / TPB, TPB, 0, stream>>>(acc, NC);
  float* cur = hA;
  float* nxt = hB;
  long long ethreads = (long long)E * 32;
  int eblocks = (int)((ethreads + TPB - 1) / TPB);
  int ckpt = 0;
  for (int step = 1; step <= 8; ++step) {
    k_prop_self <<<(NC + TPB - 1) / TPB, TPB, 0, stream>>>(cur, deg, nxt, N);
    k_prop_edges<<<eblocks, TPB, 0, stream>>>(src, dst, deg, cur, nxt, E);
    float* t = cur; cur = nxt; nxt = t;
    if (step == 1 || step == 2 || step == 4 || step == 8) {
      k_accum<<<(NC + TPB - 1) / TPB, TPB, 0, stream>>>(acc, cur, att_s + ckpt * NB_DIM, N);
      ++ckpt;
    }
  }

  // inverse rotation + residual, in place on d_out
  long long fthreads = (long long)N * NB_DIM;
  k_final<<<(int)((fthreads + TPB - 1) / TPB), TPB, 0, stream>>>(x, cth, sth, acc, N);
}
